// CardioformerMamba_69930657513947
// MI455X (gfx1250) — matrix operand order repaired and checked by reading, not compiled, not run on hardware
//
#include <hip/hip_runtime.h>
#include <hip/hip_bf16.h>

typedef __attribute__((ext_vector_type(16))) __bf16 v16bf;
typedef __attribute__((ext_vector_type(8)))  __bf16 v8bf;
typedef __attribute__((ext_vector_type(2)))  __bf16 v2bf;
typedef __attribute__((ext_vector_type(8)))  float  v8f;

// ---------- helpers ----------
// pack two f32 -> packed 2xbf16 (one dword).
__device__ __forceinline__ unsigned int pack2bf(float a, float b) {
#if __has_builtin(__builtin_amdgcn_cvt_pk_bf16_f32)
  v2bf p = __builtin_amdgcn_cvt_pk_bf16_f32(a, b);
  return __builtin_bit_cast(unsigned int, p);
#else
  // round-half-up then take high 16 bits of each; single v_perm_b32 packs both.
  unsigned int ua = __builtin_bit_cast(unsigned int, a) + 0x8000u;
  unsigned int ub = __builtin_bit_cast(unsigned int, b) + 0x8000u;
  // perm: src1 (ua) = bytes 0..3, src0 (ub) = bytes 4..7
  return __builtin_amdgcn_perm(ub, ua, 0x07060302u);
#endif
}

__device__ __forceinline__ v16bf cat8(v8bf lo, v8bf hi) {
  return __builtin_shufflevector(lo, hi, 0,1,2,3,4,5,6,7,8,9,10,11,12,13,14,15);
}

#define TILE_M 128
#define TILE_N 128
#define TILE_K 32
#define LDS_LD 40   // ushorts per row: 32 data + 8 pad (80B, 16B aligned)

// A fragment (16x32 bf16): lane&15 = row M.
// lane<16 -> elems 0..7 = K0..7, 8..15 = K16..23; lane>=16 -> +8.
__device__ __forceinline__ v16bf load_afrag(const unsigned short* As, int rowBase, int lane) {
  int m    = rowBase + (lane & 15);
  int half = lane >> 4;
  const unsigned short* p = As + m * LDS_LD + half * 8;
  v8bf lo = *reinterpret_cast<const v8bf*>(p);
  v8bf hi = *reinterpret_cast<const v8bf*>(p + 16);
  return cat8(lo, hi);
}

// B fragment (32x16 bf16): lane&15 = col N; lane<16 -> elems 0..7 = K0..7, 8..15 = K16..23; lane>=16 -> +8.
__device__ __forceinline__ v16bf load_bfrag(const unsigned short* Bs, int colBase, int lane) {
  int n    = colBase + (lane & 15);
  int half = lane >> 4;
  const unsigned short* p = Bs + n * LDS_LD + half * 8;
  v8bf lo = *reinterpret_cast<const v8bf*>(p);
  v8bf hi = *reinterpret_cast<const v8bf*>(p + 16);
  return cat8(lo, hi);
}

// ---------- generic WMMA GEMM:  C[M,N] = A[M,K(lda)] * W[N,K]^T  (+ epilogue) ----------
// MODE 0: none.  MODE 1: v = softplus(v + bias[n])
// EDGE:  true -> bounds-check tile loads/stores (only needed for N=96 GEMM)
template <bool EDGE, int MODE>
__global__ __launch_bounds__(256)
void gemm_bf16_wmma(const float* __restrict__ A, const float* __restrict__ W,
                    float* __restrict__ C, int M, int N, int K, int lda,
                    const float* __restrict__ bias) {
  __shared__ __align__(16) unsigned short As[TILE_M * LDS_LD];
  __shared__ __align__(16) unsigned short Bs[TILE_N * LDS_LD];

  const int tid  = threadIdx.x;
  const int lane = tid & 31;
  const int wave = tid >> 5;
  const int wm   = wave & 3;   // 4 row groups of 32
  const int wn   = wave >> 2;  // 2 col groups of 64
  const int m0   = blockIdx.y * TILE_M;
  const int n0   = blockIdx.x * TILE_N;

  // this thread's fixed staging coordinates: rows r, r+32, r+64, r+96 ; cols c4..c4+3
  const int r  = tid >> 3;
  const int c4 = (tid & 7) << 2;

  v8f acc[2][4] = {};
  float4 ra[4], rb[4];

  const int kTiles = K / TILE_K;

  auto issue_loads = [&](int k0) {
    #pragma unroll
    for (int s = 0; s < 4; ++s) {
      int gm = m0 + r + 32 * s;
      int gn = n0 + r + 32 * s;
      if (EDGE) {
        ra[s] = make_float4(0.f, 0.f, 0.f, 0.f);
        rb[s] = make_float4(0.f, 0.f, 0.f, 0.f);
        if (gm < M) ra[s] = *reinterpret_cast<const float4*>(A + (size_t)gm * lda + k0 + c4);
        if (gn < N) rb[s] = *reinterpret_cast<const float4*>(W + (size_t)gn * K + k0 + c4);
      } else {
        ra[s] = *reinterpret_cast<const float4*>(A + (size_t)gm * lda + k0 + c4);
        rb[s] = *reinterpret_cast<const float4*>(W + (size_t)gn * K + k0 + c4);
      }
    }
  };

  issue_loads(0);

  for (int kt = 0; kt < kTiles; ++kt) {
    // convert staged registers -> bf16 LDS tiles (one v_perm_b32 per dword)
    #pragma unroll
    for (int s = 0; s < 4; ++s) {
      unsigned int* da = reinterpret_cast<unsigned int*>(&As[(r + 32 * s) * LDS_LD + c4]);
      da[0] = pack2bf(ra[s].x, ra[s].y);
      da[1] = pack2bf(ra[s].z, ra[s].w);
      unsigned int* db = reinterpret_cast<unsigned int*>(&Bs[(r + 32 * s) * LDS_LD + c4]);
      db[0] = pack2bf(rb[s].x, rb[s].y);
      db[1] = pack2bf(rb[s].z, rb[s].w);
    }
    __syncthreads();

    // start next tile's global loads while WMMAs run on this tile
    if (kt + 1 < kTiles) issue_loads((kt + 1) * TILE_K);
    // light prefetch two tiles ahead (global_prefetch_b8)
    if (kt + 2 < kTiles) {
      int k2 = (kt + 2) * TILE_K;
      if (!EDGE || (m0 + r < M)) __builtin_prefetch(A + (size_t)(m0 + r) * lda + k2 + c4, 0, 0);
      if (!EDGE || (n0 + r < N)) __builtin_prefetch(W + (size_t)(n0 + r) * K + k2 + c4, 0, 0);
    }

    v16bf af[2];
    #pragma unroll
    for (int i = 0; i < 2; ++i) af[i] = load_afrag(As, wm * 32 + i * 16, lane);
    #pragma unroll
    for (int j = 0; j < 4; ++j) {
      v16bf bf = load_bfrag(Bs, wn * 64 + j * 16, lane);
      #pragma unroll
      for (int i = 0; i < 2; ++i) {
        acc[i][j] = __builtin_amdgcn_wmma_f32_16x16x32_bf16(
            false, af[i], false, bf, (short)0, acc[i][j], false, false);
      }
    }
    __syncthreads();
  }

  // store C (16x16 f32 D layout: VGPR r -> row r + 8*half, col = lane&15)
  const int half  = lane >> 4;
  const int nlane = lane & 15;
  #pragma unroll
  for (int i = 0; i < 2; ++i) {
    #pragma unroll
    for (int j = 0; j < 4; ++j) {
      #pragma unroll
      for (int rr = 0; rr < 8; ++rr) {
        int gm = m0 + wm * 32 + i * 16 + half * 8 + rr;
        int gn = n0 + wn * 64 + j * 16 + nlane;
        if (!EDGE || (gm < M && gn < N)) {
          float v = acc[i][j][rr];
          if (MODE == 1) {
            v += bias[gn];
            v = (v > 20.f) ? v : log1pf(__expf(v));   // softplus
          }
          C[(size_t)gm * N + gn] = v;
        }
      }
    }
  }
}

// ---------- depthwise causal conv (k=4) + SiLU ----------
__global__ void conv_silu_kernel(const float* __restrict__ xz,
                                 const float* __restrict__ conv_w,
                                 const float* __restrict__ conv_b,
                                 float* __restrict__ u,
                                 int Ltot, int L, int Din, int ldxz) {
  int idx = blockIdx.x * blockDim.x + threadIdx.x;   // over (B*L)*Din
  if (idx >= Ltot * Din) return;
  int d  = idx % Din;
  int bl = idx / Din;
  int l  = bl % L;
  float acc = conv_b[d];
  #pragma unroll
  for (int j = 0; j < 4; ++j) {
    int ls = l - 3 + j;
    if (ls >= 0) acc += conv_w[d * 4 + j] * xz[(size_t)(bl - l + ls) * ldxz + d];
  }
  u[(size_t)bl * Din + d] = acc / (1.f + __expf(-acc));   // SiLU
}

// ---------- selective scan: one (b,d) channel per thread ----------
// Single-wave (32-thread) workgroups: barrier ops degrade to S_NOP, LDS
// broadcast stays wave-internal, so the serial L loop has zero sync overhead.
__global__ __launch_bounds__(32)
void scan_kernel(const float* __restrict__ delta, const float* __restrict__ u,
                 const float* __restrict__ xdbl, const float* __restrict__ xz,
                 const float* __restrict__ A_log, const float* __restrict__ D_param,
                 float* __restrict__ y,
                 int L, int Din, int ldxz, int ldxd) {
  const int d = blockIdx.x * 32 + threadIdx.x;
  const int b = blockIdx.y;
  __shared__ float sBC[32];   // B[16] then C[16] for current (b,l)

  float Arow[16];
  #pragma unroll
  for (int n = 0; n < 16; ++n) Arow[n] = -__expf(A_log[d * 16 + n]);
  const float Dp = D_param[d];

  float st[16];
  #pragma unroll
  for (int n = 0; n < 16; ++n) st[n] = 0.f;

  for (int l = 0; l < L; ++l) {
    const size_t row = (size_t)(b * L + l);
    sBC[threadIdx.x] = xdbl[row * ldxd + 64 + threadIdx.x];
    __syncthreads();   // single-wave WG: S_NOP + LDS wait only

    const float dlt = delta[row * Din + d];
    const float uu  = u[row * Din + d];
    const float dbu = dlt * uu;
    float yv = 0.f;
    #pragma unroll
    for (int n = 0; n < 16; ++n) {
      float dA = __expf(dlt * Arow[n]);
      st[n] = dA * st[n] + dbu * sBC[n];
      yv += st[n] * sBC[16 + n];
    }
    yv += uu * Dp;
    const float z = xz[row * ldxz + Din + d];
    yv *= z / (1.f + __expf(-z));          // * SiLU(z)
    y[row * Din + d] = yv;
    __syncthreads();
  }
}

// ---------- launch ----------
extern "C" void kernel_launch(void* const* d_in, const int* in_sizes, int n_in,
                              void* d_out, int out_size, void* d_ws, size_t ws_size,
                              hipStream_t stream) {
  const float* x       = (const float*)d_in[0];
  const float* W_in    = (const float*)d_in[1];
  const float* conv_w  = (const float*)d_in[2];
  const float* conv_b  = (const float*)d_in[3];
  const float* W_x     = (const float*)d_in[4];
  const float* W_dt    = (const float*)d_in[5];
  const float* b_dt    = (const float*)d_in[6];
  const float* A_log   = (const float*)d_in[7];
  const float* D_param = (const float*)d_in[8];
  const float* W_out   = (const float*)d_in[9];
  float* out = (float*)d_out;

  constexpr int Bb = 4, Ll = 512, Dm = 1024, Din = 2048, Nst = 16, Rdt = 64;
  constexpr int M = Bb * Ll;              // 2048 rows
  constexpr int Nxd = Rdt + 2 * Nst;      // 96

  float* xz    = (float*)d_ws;                       // M x 4096
  float* u     = xz    + (size_t)M * 2 * Din;        // M x 2048
  float* xdbl  = u     + (size_t)M * Din;            // M x 96
  float* delta = xdbl  + (size_t)M * Nxd;            // M x 2048
  float* y     = delta + (size_t)M * Din;            // M x 2048

  dim3 blk(256);

  // 1) xz = x @ W_in^T          (2048 x 4096, K=1024) : interior only
  gemm_bf16_wmma<false, 0><<<dim3((2 * Din) / TILE_N, M / TILE_M), blk, 0, stream>>>(
      x, W_in, xz, M, 2 * Din, Dm, Dm, nullptr);

  // 2) depthwise causal conv + SiLU -> u
  {
    int tot = M * Din;
    conv_silu_kernel<<<(tot + 255) / 256, blk, 0, stream>>>(xz, conv_w, conv_b, u, M, Ll, Din, 2 * Din);
  }

  // 3) x_dbl = u @ W_x^T        (2048 x 96, K=2048) : N edge -> guarded
  gemm_bf16_wmma<true, 0><<<dim3((Nxd + TILE_N - 1) / TILE_N, M / TILE_M), blk, 0, stream>>>(
      u, W_x, xdbl, M, Nxd, Din, Din, nullptr);

  // 4) delta = softplus(dlt @ W_dt^T + b_dt)   (2048 x 2048, K=64; A = x_dbl[:, :64], lda=96)
  gemm_bf16_wmma<false, 1><<<dim3(Din / TILE_N, M / TILE_M), blk, 0, stream>>>(
      xdbl, W_dt, delta, M, Din, Rdt, Nxd, b_dt);

  // 5) selective scan (+ D skip + SiLU(z) gate) -> y : single-wave blocks
  scan_kernel<<<dim3(Din / 32, Bb), dim3(32), 0, stream>>>(
      delta, u, xdbl, xz, A_log, D_param, y, Ll, Din, 2 * Din, Nxd);

  // 6) out = y @ W_out^T        (2048 x 1024, K=2048) : interior only
  gemm_bf16_wmma<false, 0><<<dim3(Dm / TILE_N, M / TILE_M), blk, 0, stream>>>(
      y, W_out, out, M, Dm, Din, Din, nullptr);
}
